// VGG16RoIHead_Side2_10728828305971
// MI455X (gfx1250) — matrix operand order repaired and checked by reading, not compiled, not run on hardware
//
#include <hip/hip_runtime.h>
#include <hip/hip_bf16.h>

// ---------------------------------------------------------------------------
// MI455X (gfx1250, wave32) implementation.
// GEMMs run on v_wmma_f32_16x16x32_bf16: fp32 operands are streamed from HBM
// (the workload is weight-bandwidth bound: ~592MB @ 23.3TB/s ~ 25us floor),
// converted to bf16 in-register (v_cvt_pk_bf16_f32), accumulated in fp32.
// Block tile 128x64, 8 waves, 32x32 per wave -> 4 WMMAs per K-step.
// ---------------------------------------------------------------------------

typedef __attribute__((ext_vector_type(16))) __bf16 v16bf;
typedef __attribute__((ext_vector_type(8)))  __bf16 v8bf;
typedef __attribute__((ext_vector_type(2)))  __bf16 v2bf;
typedef __attribute__((ext_vector_type(8)))  float  v8f;

#define FEAT_C 512
#define FEAT_H 50
#define FEAT_W 75
#define FEAT_HW 3750
#define NROI 256

// ----------------------------- WMMA GEMM -----------------------------------
// C[M,N] = relu(A[M,K] @ B[K,N] + bias), row-major, M%128==0, N%64==0, K%32==0.
// 256 threads = 8 waves arranged 4(M) x 2(N); each wave owns a 32x32 strip:
// 2 A-frags x 2 B-frags -> 4 f32 accumulators, 4 WMMAs per 32-K step.
// LDS: A tile [m][k] row-major bf16 (fills are 1 ds_store_b128 per 8 values);
// B tile stored transposed [n][k] (fills are ds_store_b32 of packed K-pairs) so
// a lane's 16 K values for a B fragment are two 16B reads (8-K chunks half and 2+half).
// Fragment layouts per CDNA5 ISA 7.12.2:
//   A: lane L: M=L%16, half=L/16; elems 0..7 -> K=half*8+j, 8..15 -> K=16+half*8+(j-8)
//   B: lane L: N=L%16, half=L/16; same K order as A
//   C/D: v8f elem i -> M=i+8*half, N=L%16
__global__ __launch_bounds__(256) void gemm_wmma_kernel(
    const float* __restrict__ A, const float* __restrict__ B,
    const float* __restrict__ bias, float* __restrict__ Cout,
    int M, int N, int K, int relu)
{
    __shared__ __bf16 lA[128 * 32];   // 8 KB
    __shared__ __bf16 lB[64 * 32];    // 4 KB

    const int tid  = threadIdx.x;
    const int lane = tid & 31;
    const int wave = tid >> 5;
    const int wm   = (wave >> 1) << 5;   // 0,32,64,96
    const int wn   = (wave & 1) << 5;    // 0,32
    const int half = lane >> 4;          // 0/1
    const int l16  = lane & 15;
    const long bm  = (long)blockIdx.y * 128;
    const long bn  = (long)blockIdx.x * 64;

    v8f acc00 = {0.f,0.f,0.f,0.f,0.f,0.f,0.f,0.f};
    v8f acc01 = acc00, acc10 = acc00, acc11 = acc00;

    for (int k0 = 0; k0 < K; k0 += 32) {
        // ---- fill A tile: 128 rows x 32 K. 512 8-wide segments, 2/thread.
        // Each segment: 2 coalesced global b128 loads -> 8 bf16 -> 1 ds_store_b128.
        #pragma unroll
        for (int r = 0; r < 2; ++r) {
            int s    = tid + r * 256;
            int m    = s >> 2;
            int koff = (s & 3) * 8;
            const float4* gp = (const float4*)&A[(bm + m) * (long)K + k0 + koff];
            float4 f0 = gp[0];
            float4 f1 = gp[1];
            v8bf pk;
            pk[0] = (__bf16)f0.x; pk[1] = (__bf16)f0.y;
            pk[2] = (__bf16)f0.z; pk[3] = (__bf16)f0.w;
            pk[4] = (__bf16)f1.x; pk[5] = (__bf16)f1.y;
            pk[6] = (__bf16)f1.z; pk[7] = (__bf16)f1.w;
            *(v8bf*)&lA[m * 32 + koff] = pk;
        }
        // ---- fill B tile transposed: 32 K x 64 N -> lB[n][k].
        // 1024 K-pair units, 4/thread: 2 coalesced global b32 loads -> ds_store_b32.
        #pragma unroll
        for (int r = 0; r < 4; ++r) {
            int u = tid + r * 256;
            int n = u & 63;
            int k = (u >> 6) * 2;
            float b0f = B[(long)(k0 + k)     * N + bn + n];
            float b1f = B[(long)(k0 + k + 1) * N + bn + n];
            v2bf pk;
            pk[0] = (__bf16)b0f;
            pk[1] = (__bf16)b1f;
            *(v2bf*)&lB[n * 32 + k] = pk;
        }
        __syncthreads();

        const v8bf* ap0 = (const v8bf*)&lA[(wm + l16) * 32];
        const v8bf* ap1 = (const v8bf*)&lA[(wm + 16 + l16) * 32];
        v16bf afrag0 = __builtin_shufflevector(ap0[half], ap0[2 + half],
            0,1,2,3,4,5,6,7,8,9,10,11,12,13,14,15);
        v16bf afrag1 = __builtin_shufflevector(ap1[half], ap1[2 + half],
            0,1,2,3,4,5,6,7,8,9,10,11,12,13,14,15);

        const v8bf* bp0 = (const v8bf*)&lB[(wn + l16) * 32];
        const v8bf* bp1 = (const v8bf*)&lB[(wn + 16 + l16) * 32];
        v16bf bfrag0 = __builtin_shufflevector(bp0[half], bp0[2 + half],
            0,1,2,3,4,5,6,7,8,9,10,11,12,13,14,15);
        v16bf bfrag1 = __builtin_shufflevector(bp1[half], bp1[2 + half],
            0,1,2,3,4,5,6,7,8,9,10,11,12,13,14,15);

        acc00 = __builtin_amdgcn_wmma_f32_16x16x32_bf16(
            false, afrag0, false, bfrag0, (short)0, acc00, false, false);
        acc01 = __builtin_amdgcn_wmma_f32_16x16x32_bf16(
            false, afrag0, false, bfrag1, (short)0, acc01, false, false);
        acc10 = __builtin_amdgcn_wmma_f32_16x16x32_bf16(
            false, afrag1, false, bfrag0, (short)0, acc10, false, false);
        acc11 = __builtin_amdgcn_wmma_f32_16x16x32_bf16(
            false, afrag1, false, bfrag1, (short)0, acc11, false, false);
        __syncthreads();
    }

    const long gn0 = bn + wn + l16;
    const long gn1 = gn0 + 16;
    const float bv0 = bias[gn0];
    const float bv1 = bias[gn1];
    #pragma unroll
    for (int i = 0; i < 8; ++i) {
        long gm0 = bm + wm + i + half * 8;
        long gm1 = gm0 + 16;
        float v00 = acc00[i] + bv0;
        float v01 = acc01[i] + bv1;
        float v10 = acc10[i] + bv0;
        float v11 = acc11[i] + bv1;
        if (relu) {
            v00 = fmaxf(v00, 0.f); v01 = fmaxf(v01, 0.f);
            v10 = fmaxf(v10, 0.f); v11 = fmaxf(v11, 0.f);
        }
        Cout[gm0 * N + gn0] = v00;
        Cout[gm0 * N + gn1] = v01;
        Cout[gm1 * N + gn0] = v10;
        Cout[gm1 * N + gn1] = v11;
    }
}

// --------------------------- ROI align -------------------------------------
// One block per output row (ROI); boxes are rows of 4 floats (x1,y1,x2,y2).
// Output row layout [c][py][px], row length 512*ph*pw. sr=2 sampling, mean.
__global__ __launch_bounds__(256) void roi_align_kernel(
    const float* __restrict__ feat, const float* __restrict__ boxes,
    float* __restrict__ out, int ph, int pw)
{
    const int row = blockIdx.x;
    const float bx1 = boxes[row * 4 + 0];
    const float by1 = boxes[row * 4 + 1];
    const float bx2 = boxes[row * 4 + 2];
    const float by2 = boxes[row * 4 + 3];
    const float sc  = 1.f / 16.f;
    const float rx1 = bx1 * sc, ry1 = by1 * sc;
    const float rx2 = bx2 * sc, ry2 = by2 * sc;
    const float bw = fmaxf(rx2 - rx1, 1.f) / (float)pw;
    const float bh = fmaxf(ry2 - ry1, 1.f) / (float)ph;
    const int nb  = ph * pw;
    const int tot = FEAT_C * nb;

    for (int idx = threadIdx.x; idx < tot; idx += 256) {
        int px = idx % pw;
        int py = (idx / pw) % ph;
        int c  = idx / nb;
        const float* f = feat + (size_t)c * FEAT_HW;
        float acc = 0.f;
        #pragma unroll
        for (int iy = 0; iy < 2; ++iy) {
            float yy = ry1 + ((float)py + (iy + 0.5f) * 0.5f) * bh;
            bool vy = (yy >= -1.f) && (yy <= (float)FEAT_H);
            float y = fminf(fmaxf(yy, 0.f), (float)(FEAT_H - 1));
            float y0f = floorf(y);
            int y0 = (int)y0f;
            int y1 = (y0 + 1 > FEAT_H - 1) ? (FEAT_H - 1) : (y0 + 1);
            float ly = y - y0f;
            #pragma unroll
            for (int ix = 0; ix < 2; ++ix) {
                float xx = rx1 + ((float)px + (ix + 0.5f) * 0.5f) * bw;
                bool vx = (xx >= -1.f) && (xx <= (float)FEAT_W);
                float x = fminf(fmaxf(xx, 0.f), (float)(FEAT_W - 1));
                float x0f = floorf(x);
                int x0 = (int)x0f;
                int x1 = (x0 + 1 > FEAT_W - 1) ? (FEAT_W - 1) : (x0 + 1);
                float lx = x - x0f;
                float v = (1.f - ly) * (1.f - lx) * f[y0 * FEAT_W + x0]
                        + (1.f - ly) * lx         * f[y0 * FEAT_W + x1]
                        + ly * (1.f - lx)         * f[y1 * FEAT_W + x0]
                        + ly * lx                 * f[y1 * FEAT_W + x1];
                acc += (vy && vx) ? v : 0.f;
            }
        }
        out[(size_t)row * tot + idx] = acc * 0.25f;
    }
}

// ------------------------ side box geometry ---------------------------------
// rois rows = [y1, x1, y2, x2]; produce 4 clipped side boxes (x1,y1,x2,y2).
__global__ void side_boxes_kernel(const float* __restrict__ rois,
                                  float* __restrict__ sbox)
{
    int i = blockIdx.x * blockDim.x + threadIdx.x;
    if (i >= NROI) return;
    float y1 = rois[i * 4 + 0], x1 = rois[i * 4 + 1];
    float y2 = rois[i * 4 + 2], x2 = rois[i * 4 + 3];
    float dw = x2 - x1, dh = y2 - y1;
    float sw = fmaxf(2.0f * log2f(dw), 0.2f * dw);
    float sh = fmaxf(2.0f * log2f(dh), 0.2f * dh);
    const float cmax = 1200.0f;  // int(W / SCALE) = 75*16
    auto clip = [cmax](float v) { return fminf(fmaxf(v, 0.f), cmax); };
    // side 0: left   (x1-0.5sw, y1, x1, y2)
    float* p = sbox + 0 * NROI * 4 + i * 4;
    p[0] = clip(x1 - 0.5f * sw); p[1] = clip(y1); p[2] = clip(x1); p[3] = clip(y2);
    // side 1: right  (x2-0.5sw, y1, x2+0.5sw, y2)
    p = sbox + 1 * NROI * 4 + i * 4;
    p[0] = clip(x2 - 0.5f * sw); p[1] = clip(y1); p[2] = clip(x2 + 0.5f * sw); p[3] = clip(y2);
    // side 2: top    (x1, y1-0.5sh, x2, y1)
    p = sbox + 2 * NROI * 4 + i * 4;
    p[0] = clip(x1); p[1] = clip(y1 - 0.5f * sh); p[2] = clip(x2); p[3] = clip(y1);
    // side 3: bottom (x1, y2-0.5sh, x2, y2+0.5sh)
    p = sbox + 3 * NROI * 4 + i * 4;
    p[0] = clip(x1); p[1] = clip(y2 - 0.5f * sh); p[2] = clip(x2); p[3] = clip(y2 + 0.5f * sh);
}

// ------------------ per-ROI side-delta GEMVs (2048-dot) ---------------------
__global__ __launch_bounds__(256) void delta_kernel(
    const float* __restrict__ fc,
    const float* __restrict__ w0, const float* __restrict__ b0,
    const float* __restrict__ w1, const float* __restrict__ b1,
    const float* __restrict__ w2, const float* __restrict__ b2,
    const float* __restrict__ w3, const float* __restrict__ b3,
    float* __restrict__ dvals)
{
    int roi = blockIdx.x, head = blockIdx.y;
    const float* w = (head == 0) ? w0 : (head == 1) ? w1 : (head == 2) ? w2 : w3;
    const float* row = fc + (size_t)(head * NROI + roi) * 2048;
    float s = 0.f;
    for (int k = threadIdx.x; k < 2048; k += 256) s = fmaf(row[k], w[k], s);
    __shared__ float red[256];
    red[threadIdx.x] = s;
    __syncthreads();
    for (int off = 128; off > 0; off >>= 1) {
        if (threadIdx.x < off) red[threadIdx.x] += red[threadIdx.x + off];
        __syncthreads();
    }
    if (threadIdx.x == 0) {
        const float* bb = (head == 0) ? b0 : (head == 1) ? b1 : (head == 2) ? b2 : b3;
        dvals[head * NROI + roi] = red[0] + bb[0];
    }
}

// ---------------- fine boxes + delta/box outputs ----------------------------
__global__ void fine_kernel(const float* __restrict__ rois,
                            const float* __restrict__ dvals,
                            float* __restrict__ fbox,
                            float* __restrict__ out_deltas,   // (256,4) [dl,dr,dt,db]
                            float* __restrict__ out_boxes)    // (256,4) [y1f,x1f,y2f,x2f]
{
    int i = blockIdx.x * blockDim.x + threadIdx.x;
    if (i >= NROI) return;
    float y1 = rois[i * 4 + 0], x1 = rois[i * 4 + 1];
    float y2 = rois[i * 4 + 2], x2 = rois[i * 4 + 3];
    float dl = dvals[0 * NROI + i], dr = dvals[1 * NROI + i];
    float dt = dvals[2 * NROI + i], db = dvals[3 * NROI + i];
    const float std = 0.1f;
    float x1f = dl * std * (x2 - x1) + x1;
    float y1f = dt * std * (y2 - y1) + y1;
    float x2f = dr * std * (x2 - x1f) + x2;
    float y2f = db * std * (y2 - y1f) + y2;
    fbox[i * 4 + 0] = x1f; fbox[i * 4 + 1] = y1f;
    fbox[i * 4 + 2] = x2f; fbox[i * 4 + 3] = y2f;
    out_deltas[i * 4 + 0] = dl; out_deltas[i * 4 + 1] = dr;
    out_deltas[i * 4 + 2] = dt; out_deltas[i * 4 + 3] = db;
    out_boxes[i * 4 + 0] = y1f; out_boxes[i * 4 + 1] = x1f;
    out_boxes[i * 4 + 2] = y2f; out_boxes[i * 4 + 3] = x2f;
}

// ------------- boundary concat: (256 x 8192) from fc blocks -----------------
__global__ void catb_kernel(const float* __restrict__ fc, float* __restrict__ cat)
{
    int idx = blockIdx.x * blockDim.x + threadIdx.x;
    if (idx >= NROI * 8192) return;
    int i  = idx >> 13;          // row
    int j  = idx & 8191;
    int s  = j >> 11;            // which side block
    int jj = j & 2047;
    cat[idx] = fc[(size_t)(s * NROI + i) * 2048 + jj];
}

// ------------------- scores: (256 x 21), K = 4096 + 1024 --------------------
__global__ __launch_bounds__(64) void scores_kernel(
    const float* __restrict__ fcen, const float* __restrict__ fbnd,
    const float* __restrict__ w, const float* __restrict__ b,
    float* __restrict__ out)
{
    int i = blockIdx.x;
    int j = threadIdx.x;
    if (j >= 21) return;
    float s = b[j];
    const float* fi = fcen + (size_t)i * 4096;
    for (int k = 0; k < 4096; ++k) s = fmaf(fi[k], w[(size_t)k * 21 + j], s);
    const float* fb = fbnd + (size_t)i * 1024;
    for (int k = 0; k < 1024; ++k) s = fmaf(fb[k], w[(size_t)(4096 + k) * 21 + j], s);
    out[i * 21 + j] = s;
}

// ---------------------------------------------------------------------------
extern "C" void kernel_launch(void* const* d_in, const int* in_sizes, int n_in,
                              void* d_out, int out_size, void* d_ws, size_t ws_size,
                              hipStream_t stream)
{
    (void)in_sizes; (void)n_in; (void)out_size; (void)ws_size;
    const float* x     = (const float*)d_in[1];   // (1,512,50,75); x0 unused
    const float* rois  = (const float*)d_in[2];   // (256,4) [y1,x1,y2,x2]
    const float* c1_w1 = (const float*)d_in[4];
    const float* c1_b1 = (const float*)d_in[5];
    const float* c1_w2 = (const float*)d_in[6];
    const float* c1_b2 = (const float*)d_in[7];
    const float* c2_w  = (const float*)d_in[8];
    const float* c2_b  = (const float*)d_in[9];
    const float* se_w1 = (const float*)d_in[10];
    const float* se_b1 = (const float*)d_in[11];
    const float* se_w2 = (const float*)d_in[12];
    const float* se_b2 = (const float*)d_in[13];
    const float* ll_w  = (const float*)d_in[14];
    const float* ll_b  = (const float*)d_in[15];
    const float* lr_w  = (const float*)d_in[16];
    const float* lr_b  = (const float*)d_in[17];
    const float* lt_w  = (const float*)d_in[18];
    const float* lt_b  = (const float*)d_in[19];
    const float* lb_w  = (const float*)d_in[20];
    const float* lb_b  = (const float*)d_in[21];
    const float* sc_w  = (const float*)d_in[22];
    const float* sc_b  = (const float*)d_in[23];
    float* out = (float*)d_out;   // [0,1024): deltas | [1024,6400): scores | [6400,7424): boxes

    // workspace layout (floats), aliased where lifetimes allow (~54 MB total)
    float* ws    = (float*)d_ws;
    float* sbox  = ws;                    // 4*256*4  = 4096
    float* fbox  = sbox + 4096;           // 256*4    = 1024
    float* dvals = fbox + 1024;           // 4*256    = 1024
    float* fc    = dvals + 1024;          // 1024*2048 = 2097152
    float* f_cen = fc + 2097152;          // 256*4096 = 1048576
    float* f_bnd = f_cen + 1048576;       // 256*1024 = 262144
    float* bufB  = f_bnd + 262144;        // 1024*2048 (hid_se / hid_c / cat_b)
    float* bufA  = bufB + 2097152;        // 1024*7680 (pooled sides / pooled center)

    // 1) side boxes, 2) side ROI-aligns -> (1024 x 7680), rows [L;R;T;B]
    side_boxes_kernel<<<1, 256, 0, stream>>>(rois, sbox);
    for (int s = 0; s < 4; ++s) {
        int ph = (s < 2) ? 3 : 5;
        int pw = (s < 2) ? 5 : 3;
        roi_align_kernel<<<NROI, 256, 0, stream>>>(
            x, sbox + (size_t)s * NROI * 4, bufA + (size_t)s * NROI * 7680, ph, pw);
    }

    // 3) side MLP: (1024x7680)@(7680x2048) relu, then @(2048x2048) relu -> fc
    gemm_wmma_kernel<<<dim3(2048 / 64, 1024 / 128), 256, 0, stream>>>(
        bufA, se_w1, se_b1, bufB, 1024, 2048, 7680, 1);
    gemm_wmma_kernel<<<dim3(2048 / 64, 1024 / 128), 256, 0, stream>>>(
        bufB, se_w2, se_b2, fc, 1024, 2048, 2048, 1);

    // 4) per-side deltas + fine boxes (+ write deltas/boxes outputs)
    delta_kernel<<<dim3(NROI, 4), 256, 0, stream>>>(
        fc, ll_w, ll_b, lr_w, lr_b, lt_w, lt_b, lb_w, lb_b, dvals);
    fine_kernel<<<1, 256, 0, stream>>>(rois, dvals, fbox, out, out + 6400);

    // 5) center ROI-align 7x7 -> (256 x 25088)
    roi_align_kernel<<<NROI, 256, 0, stream>>>(x, fbox, bufA, 7, 7);

    // 6) center MLP: @(25088x4096) relu, @(4096x4096) relu -> f_center
    gemm_wmma_kernel<<<dim3(4096 / 64, 256 / 128), 256, 0, stream>>>(
        bufA, c1_w1, c1_b1, bufB, 256, 4096, 25088, 1);
    gemm_wmma_kernel<<<dim3(4096 / 64, 256 / 128), 256, 0, stream>>>(
        bufB, c1_w2, c1_b2, f_cen, 256, 4096, 4096, 1);

    // 7) boundary branch: concat fc blocks along features, @(8192x1024) relu
    catb_kernel<<<(NROI * 8192) / 256, 256, 0, stream>>>(fc, bufB);
    gemm_wmma_kernel<<<dim3(1024 / 64, 256 / 128), 256, 0, stream>>>(
        bufB, c2_w, c2_b, f_bnd, 256, 1024, 8192, 1);

    // 8) scores: [f_center | f_boundary] @ sc_w + sc_b -> (256 x 21)
    scores_kernel<<<NROI, 64, 0, stream>>>(f_cen, f_bnd, sc_w, sc_b, out + 1024);
}
